// PointerNetworks_61160334295695
// MI455X (gfx1250) — compile-verified, matrix operand order checked by reading, not run on hardware
//
#include <hip/hip_runtime.h>
#include <math.h>

// ---------------------------------------------------------------------------
// Pointer network (bi-GRU encoder + GRU decoder + pointer attention) for
// MI455X / gfx1250, wave32, WMMA f32_16x16x32_f16 + async global->LDS DMA.
// ---------------------------------------------------------------------------

typedef __attribute__((ext_vector_type(16))) _Float16 v16h;
typedef __attribute__((ext_vector_type(8)))  float    v8f;
typedef __attribute__((ext_vector_type(4)))  int      v4i_t;

#define BB 32
#define LL 1024
#define WDIM 1024
#define HH 512
#define DD 1024
#define KK 128

// Async global->LDS DMA (gfx1250): guarded so we fall back to register-staged
// copies if this toolchain doesn't expose the builtins.
#if __has_builtin(__builtin_amdgcn_global_load_async_to_lds_b128)
#define PTR_ASYNC_LDS 1
#else
#define PTR_ASYNC_LDS 0
#endif

// builtin params are typed v4i pointers: global (AS1) src, LDS (AS3) dst
#define PTR_GLB(p) ((__attribute__((address_space(1))) v4i_t*)(p))
#define PTR_LDS(p) ((__attribute__((address_space(3))) v4i_t*)(p))

__device__ inline void ptr_async_fence() {
#if PTR_ASYNC_LDS
#if __has_builtin(__builtin_amdgcn_s_wait_asynccnt)
  __builtin_amdgcn_s_wait_asynccnt(0);
#else
  asm volatile("s_wait_asynccnt 0x0" ::: "memory");
#endif
#endif
}

union FragU { v16h v; uint4 q[2]; };

// WMMA 16-bit A/B fragment loader for a 16x32 (rows x k) tile, row-major
// source with leading dimension ld (halves). Layout per CDNA5 ISA 7.12.2:
//   lanes 0-15 : row = lane,    halves 0..7 = k..k+7,    halves 8..15 = k+16..k+23
//   lanes 16-31: row = lane-16, halves 0..7 = k+8..k+15, halves 8..15 = k+24..k+31
__device__ inline v16h ldfrag(const _Float16* __restrict__ p, int rowbase, int ld, int kk) {
  const int lane = threadIdx.x & 31;
  const int r    = lane & 15;
  const int ks   = (lane >> 4) << 3;           // 0 or 8
  const _Float16* b = p + (size_t)(rowbase + r) * ld + kk + ks;
  FragU f;
  f.q[0] = *(const uint4*)(b);
  f.q[1] = *(const uint4*)(b + 16);
  return f.v;
}

__device__ inline v8f wmma16(v16h a, v16h b, v8f c) {
  return __builtin_amdgcn_wmma_f32_16x16x32_f16(false, a, false, b, (short)0, c, false, false);
}

// ---------------------------------------------------------------------------
// Elementwise helpers
// ---------------------------------------------------------------------------
__global__ void ptr_f32_to_f16(const float* __restrict__ in, _Float16* __restrict__ out, int n) {
  int i = blockIdx.x * 256 + threadIdx.x;
  if (i < n) out[i] = (_Float16)in[i];
}

__global__ void ptr_zero_f32(float* __restrict__ p, int n) {
  int i = blockIdx.x * 256 + threadIdx.x;
  if (i < n) p[i] = 0.0f;
}

__global__ void ptr_zero_f16(_Float16* __restrict__ p, int n) {
  int i = blockIdx.x * 256 + threadIdx.x;
  if (i < n) p[i] = (_Float16)0.0f;
}

// h0_dec = concat(hT_f, hT_b) in f32 and f16
__global__ void ptr_concat_h(const float* __restrict__ hf, const float* __restrict__ hb,
                             float* __restrict__ hd, _Float16* __restrict__ hd16) {
  int i = blockIdx.x * 256 + threadIdx.x;      // B*D
  if (i < BB * DD) {
    int b = i >> 10, d = i & (DD - 1);
    float v = (d < HH) ? hf[b * HH + d] : hb[b * HH + d - HH];
    hd[i]   = v;
    hd16[i] = (_Float16)v;
  }
}

__global__ void ptr_gather_decin(const _Float16* __restrict__ enc16,
                                 const int* __restrict__ Xindex,
                                 _Float16* __restrict__ decin16) {
  int bk = blockIdx.x;                          // B*K rows
  int b  = bk >> 7;                             // K = 128
  int x  = Xindex[bk];
  const _Float16* src = enc16 + ((size_t)b * LL + x) * DD;
  _Float16*       dst = decin16 + (size_t)bk * DD;
  for (int d = threadIdx.x; d < DD; d += 256) dst[d] = src[d];
}

// ---------------------------------------------------------------------------
// Generic WMMA GEMM:  C[M,N] = A[M,Kd] * W[N,Kd]^T (+ bias[N])
// block = 128 threads (4 waves, 2x2), tile 64x64, K-chunk 32.
// Double-buffered LDS; tiles staged with GLOBAL_LOAD_ASYNC_TO_LDS_B128.
// blockIdx.z batches with element strides zsA/zsW/zsC.
// ---------------------------------------------------------------------------
__global__ __launch_bounds__(128) void ptr_gemm_f16(
    const _Float16* __restrict__ A, const _Float16* __restrict__ W,
    const float* __restrict__ bias, float* __restrict__ C,
    int M, int N, int Kd,
    long long zsA, long long zsW, long long zsC)
{
  A += (long long)blockIdx.z * zsA;
  W += (long long)blockIdx.z * zsW;
  C += (long long)blockIdx.z * zsC;

  __shared__ _Float16 sA[2][64 * 32];
  __shared__ _Float16 sW[2][64 * 32];

  const int tid  = threadIdx.x;
  const int wave = tid >> 5;
  const int lane = tid & 31;
  const int wm   = (wave >> 1) * 32;            // wave row offset in block tile
  const int wn   = (wave & 1) * 32;             // wave col offset in block tile
  const int bm   = blockIdx.y * 64;
  const int bn   = blockIdx.x * 64;

  // each thread owns two 8-half chunks of each 64x32 tile
  const int c0 = tid, c1 = tid + 128;
  const int r0 = c0 >> 2, cc0 = (c0 & 3) * 8;
  const int r1 = c1 >> 2, cc1 = (c1 & 3) * 8;

  auto stage = [&](int buf, int kk) {
#if PTR_ASYNC_LDS
    __builtin_amdgcn_global_load_async_to_lds_b128(
        PTR_GLB(&A[(size_t)(bm + r0) * Kd + kk + cc0]), PTR_LDS(&sA[buf][r0 * 32 + cc0]), 0, 0);
    __builtin_amdgcn_global_load_async_to_lds_b128(
        PTR_GLB(&A[(size_t)(bm + r1) * Kd + kk + cc1]), PTR_LDS(&sA[buf][r1 * 32 + cc1]), 0, 0);
    __builtin_amdgcn_global_load_async_to_lds_b128(
        PTR_GLB(&W[(size_t)(bn + r0) * Kd + kk + cc0]), PTR_LDS(&sW[buf][r0 * 32 + cc0]), 0, 0);
    __builtin_amdgcn_global_load_async_to_lds_b128(
        PTR_GLB(&W[(size_t)(bn + r1) * Kd + kk + cc1]), PTR_LDS(&sW[buf][r1 * 32 + cc1]), 0, 0);
#else
    uint4 a0 = *(const uint4*)&A[(size_t)(bm + r0) * Kd + kk + cc0];
    uint4 a1 = *(const uint4*)&A[(size_t)(bm + r1) * Kd + kk + cc1];
    uint4 w0 = *(const uint4*)&W[(size_t)(bn + r0) * Kd + kk + cc0];
    uint4 w1 = *(const uint4*)&W[(size_t)(bn + r1) * Kd + kk + cc1];
    *(uint4*)&sA[buf][r0 * 32 + cc0] = a0;
    *(uint4*)&sA[buf][r1 * 32 + cc1] = a1;
    *(uint4*)&sW[buf][r0 * 32 + cc0] = w0;
    *(uint4*)&sW[buf][r1 * 32 + cc1] = w1;
#endif
  };

  v8f acc[2][2] = {};

  stage(0, 0);
  ptr_async_fence();
  __syncthreads();

  int buf = 0;
  for (int kk = 0; kk < Kd; kk += 32) {
    if (kk + 32 < Kd) stage(buf ^ 1, kk + 32);   // overlap next tile DMA with WMMA

    v16h a0 = ldfrag(&sA[buf][0], wm,      32, 0);
    v16h a1 = ldfrag(&sA[buf][0], wm + 16, 32, 0);
    v16h b0 = ldfrag(&sW[buf][0], wn,      32, 0);
    v16h b1 = ldfrag(&sW[buf][0], wn + 16, 32, 0);
    acc[0][0] = wmma16(a0, b0, acc[0][0]);
    acc[0][1] = wmma16(a0, b1, acc[0][1]);
    acc[1][0] = wmma16(a1, b0, acc[1][0]);
    acc[1][1] = wmma16(a1, b1, acc[1][1]);

    ptr_async_fence();
    __syncthreads();
    buf ^= 1;
  }

  // C/D layout: VGPR r -> M=r (lanes 0-15) / M=r+8 (lanes 16-31); N = lane&15
  const int n  = lane & 15;
  const int mo = (lane >> 4) * 8;
  for (int i = 0; i < 2; ++i)
    for (int j = 0; j < 2; ++j) {
      int col = bn + wn + 16 * j + n;
      float bv = bias ? bias[col] : 0.0f;
      for (int r = 0; r < 8; ++r) {
        int m = bm + wm + 16 * i + mo + r;
        C[(size_t)m * N + col] = acc[i][j][r] + bv;
      }
    }
}

// ---------------------------------------------------------------------------
// One GRU step (PyTorch gate order r,z,n), all 32 batch rows, 64 hidden cols
// per block.  h_new = (1-z)*tanh(xn + r*(Whh_n h + bhh_n)) + z*h
// h kept in f32 (carry) + f16 twin (WMMA operand, DMA'd straight into LDS).
// ---------------------------------------------------------------------------
__global__ __launch_bounds__(128) void ptr_gru_step(
    const float* __restrict__ hin,       // [32, Hc] f32 carry
    const _Float16* __restrict__ hin16,  // [32, Hc] f16 twin
    float* __restrict__ hout,            // [32, Hc]
    _Float16* __restrict__ hout16,       // [32, Hc]
    const float* __restrict__ xw,        // timestep slice, row stride xw_stride
    long long xw_stride,
    const _Float16* __restrict__ Whh,    // [3*Hc, Hc] f16
    const float* __restrict__ bhh,       // [3*Hc]
    _Float16* __restrict__ out16,        // optional f16 output slice
    long long out_stride,
    int Hc)
{
  extern __shared__ _Float16 sh[];       // 32*Hc halves (f16 copy of h)
  const int tid  = threadIdx.x;
  const int wave = tid >> 5;
  const int lane = tid & 31;

  const int halves = 32 * Hc;
  for (int i = tid * 8; i < halves; i += 128 * 8) {
#if PTR_ASYNC_LDS
    __builtin_amdgcn_global_load_async_to_lds_b128(PTR_GLB(&hin16[i]), PTR_LDS(&sh[i]), 0, 0);
#else
    *(uint4*)&sh[i] = *(const uint4*)&hin16[i];
#endif
  }
  ptr_async_fence();
  __syncthreads();

  const int j0 = blockIdx.x * 64 + wave * 16;  // this wave's 16 hidden columns

  v8f accR[2] = {};
  v8f accZ[2] = {};
  v8f accN[2] = {};
  const _Float16* Wr = Whh;
  const _Float16* Wz = Whh + (size_t)Hc * Hc;
  const _Float16* Wn = Whh + (size_t)2 * Hc * Hc;

  for (int kk = 0; kk < Hc; kk += 32) {
    v16h a0 = ldfrag(sh, 0,  Hc, kk);          // batch rows 0..15
    v16h a1 = ldfrag(sh, 16, Hc, kk);          // batch rows 16..31
    v16h br = ldfrag(Wr, j0, Hc, kk);
    v16h bz = ldfrag(Wz, j0, Hc, kk);
    v16h bn = ldfrag(Wn, j0, Hc, kk);
    accR[0] = wmma16(a0, br, accR[0]);
    accR[1] = wmma16(a1, br, accR[1]);
    accZ[0] = wmma16(a0, bz, accZ[0]);
    accZ[1] = wmma16(a1, bz, accZ[1]);
    accN[0] = wmma16(a0, bn, accN[0]);
    accN[1] = wmma16(a1, bn, accN[1]);
  }

  const int n  = lane & 15;
  const int mo = (lane >> 4) * 8;
  const int col = j0 + n;
  const float bR = bhh[col];
  const float bZ = bhh[Hc + col];
  const float bN = bhh[2 * Hc + col];

  for (int i = 0; i < 2; ++i) {
    for (int r = 0; r < 8; ++r) {
      int m = 16 * i + mo + r;                 // batch row 0..31
      const float* xrow = xw + (size_t)m * xw_stride;
      float xr = xrow[col];
      float xz = xrow[Hc + col];
      float xn = xrow[2 * Hc + col];
      float hr = accR[i][r] + bR;
      float hz = accZ[i][r] + bZ;
      float hn = accN[i][r] + bN;
      float rg = 1.0f / (1.0f + __expf(-(xr + hr)));
      float zg = 1.0f / (1.0f + __expf(-(xz + hz)));
      float ng = tanhf(xn + rg * hn);
      float ho = hin[(size_t)m * Hc + col];
      float hv = (1.0f - zg) * ng + zg * ho;
      hout[(size_t)m * Hc + col]   = hv;
      hout16[(size_t)m * Hc + col] = (_Float16)hv;
      if (out16) out16[(size_t)m * out_stride + col] = (_Float16)hv;
    }
  }
}

// ---------------------------------------------------------------------------
// Masked log-softmax NLL per (b,k) row of logits[B,K,L].
// ---------------------------------------------------------------------------
__global__ __launch_bounds__(256) void ptr_nll(
    const float* __restrict__ logits,
    const int* __restrict__ Xindex, const int* __restrict__ Yindex,
    float* __restrict__ nll)
{
  const int bk = blockIdx.x;
  const float* row = logits + (size_t)bk * LL;
  const int xi = Xindex[bk];
  const int yi = Yindex[bk];
  __shared__ float red[256];

  float mx = -3.0e38f;
  for (int l = threadIdx.x; l < LL; l += 256)
    if (l >= xi) mx = fmaxf(mx, row[l]);
  red[threadIdx.x] = mx;
  __syncthreads();
  for (int s = 128; s > 0; s >>= 1) {
    if (threadIdx.x < s) red[threadIdx.x] = fmaxf(red[threadIdx.x], red[threadIdx.x + s]);
    __syncthreads();
  }
  mx = red[0];
  __syncthreads();

  float sum = 0.0f;
  for (int l = threadIdx.x; l < LL; l += 256)
    if (l >= xi) sum += __expf(row[l] - mx);
  red[threadIdx.x] = sum;
  __syncthreads();
  for (int s = 128; s > 0; s >>= 1) {
    if (threadIdx.x < s) red[threadIdx.x] += red[threadIdx.x + s];
    __syncthreads();
  }
  if (threadIdx.x == 0) nll[bk] = -(row[yi] - mx - logf(red[0]));
}

__global__ __launch_bounds__(256) void ptr_mean(const float* __restrict__ nll,
                                                float* __restrict__ out) {
  __shared__ float red[256];
  float s = 0.0f;
  for (int i = threadIdx.x; i < BB * KK; i += 256) s += nll[i];
  red[threadIdx.x] = s;
  __syncthreads();
  for (int st = 128; st > 0; st >>= 1) {
    if (threadIdx.x < st) red[threadIdx.x] += red[threadIdx.x + st];
    __syncthreads();
  }
  if (threadIdx.x == 0) out[0] = red[0] / (float)(BB * KK);
}

// ---------------------------------------------------------------------------
// Host orchestration
// ---------------------------------------------------------------------------
extern "C" void kernel_launch(void* const* d_in, const int* in_sizes, int n_in,
                              void* d_out, int out_size, void* d_ws, size_t ws_size,
                              hipStream_t stream) {
  const float* X     = (const float*)d_in[0];
  const int*   Xidx  = (const int*)  d_in[1];
  const int*   Yidx  = (const int*)  d_in[2];
  const float* Wih_f = (const float*)d_in[3];
  const float* Whh_f = (const float*)d_in[4];
  const float* bih_f = (const float*)d_in[5];
  const float* bhh_f = (const float*)d_in[6];
  const float* Wih_b = (const float*)d_in[7];
  const float* Whh_b = (const float*)d_in[8];
  const float* bih_b = (const float*)d_in[9];
  const float* bhh_b = (const float*)d_in[10];
  const float* Wih_d = (const float*)d_in[11];
  const float* Whh_d = (const float*)d_in[12];
  const float* bih_d = (const float*)d_in[13];
  const float* bhh_d = (const float*)d_in[14];

  char* ws = (char*)d_ws;
  size_t off = 0;
  auto alloc = [&](size_t bytes) -> void* {
    void* p = ws + off;
    off = (off + bytes + 255) & ~(size_t)255;
    return p;
  };

  _Float16* X16     = (_Float16*)alloc((size_t)BB * LL * WDIM * 2);
  _Float16* Wihf16  = (_Float16*)alloc((size_t)3 * HH * WDIM * 2);
  _Float16* Whhf16  = (_Float16*)alloc((size_t)3 * HH * HH * 2);
  _Float16* Wihb16  = (_Float16*)alloc((size_t)3 * HH * WDIM * 2);
  _Float16* Whhb16  = (_Float16*)alloc((size_t)3 * HH * HH * 2);
  _Float16* Wihd16  = (_Float16*)alloc((size_t)3 * DD * DD * 2);
  _Float16* Whhd16  = (_Float16*)alloc((size_t)3 * DD * DD * 2);
  float*    xWf     = (float*)   alloc((size_t)BB * LL * 3 * HH * 4);
  float*    xWb     = (float*)   alloc((size_t)BB * LL * 3 * HH * 4);
  _Float16* enc16   = (_Float16*)alloc((size_t)BB * LL * DD * 2);
  float*    hf0     = (float*)   alloc((size_t)BB * HH * 4);
  float*    hf1     = (float*)   alloc((size_t)BB * HH * 4);
  float*    hb0     = (float*)   alloc((size_t)BB * HH * 4);
  float*    hb1     = (float*)   alloc((size_t)BB * HH * 4);
  float*    hd0     = (float*)   alloc((size_t)BB * DD * 4);
  float*    hd1     = (float*)   alloc((size_t)BB * DD * 4);
  _Float16* hf0_16  = (_Float16*)alloc((size_t)BB * HH * 2);
  _Float16* hf1_16  = (_Float16*)alloc((size_t)BB * HH * 2);
  _Float16* hb0_16  = (_Float16*)alloc((size_t)BB * HH * 2);
  _Float16* hb1_16  = (_Float16*)alloc((size_t)BB * HH * 2);
  _Float16* hd0_16  = (_Float16*)alloc((size_t)BB * DD * 2);
  _Float16* hd1_16  = (_Float16*)alloc((size_t)BB * DD * 2);
  _Float16* decin16 = (_Float16*)alloc((size_t)BB * KK * DD * 2);
  float*    xWd     = (float*)   alloc((size_t)BB * KK * 3 * DD * 4);
  _Float16* dout16  = (_Float16*)alloc((size_t)BB * KK * DD * 2);
  float*    logits  = (float*)   alloc((size_t)BB * KK * LL * 4);
  float*    nllv    = (float*)   alloc((size_t)BB * KK * 4);

  auto cvt = [&](const float* src, _Float16* dst, int n) {
    ptr_f32_to_f16<<<(n + 255) / 256, 256, 0, stream>>>(src, dst, n);
  };

  // --- f32 -> f16 conversions -------------------------------------------
  cvt(X,     X16,    BB * LL * WDIM);
  cvt(Wih_f, Wihf16, 3 * HH * WDIM);
  cvt(Whh_f, Whhf16, 3 * HH * HH);
  cvt(Wih_b, Wihb16, 3 * HH * WDIM);
  cvt(Whh_b, Whhb16, 3 * HH * HH);
  cvt(Wih_d, Wihd16, 3 * DD * DD);
  cvt(Whh_d, Whhd16, 3 * DD * DD);

  // --- input projections: xW = X * Wih^T + bih --------------------------
  {
    dim3 grid((3 * HH) / 64, (BB * LL) / 64, 1);
    ptr_gemm_f16<<<grid, 128, 0, stream>>>(X16, Wihf16, bih_f, xWf,
                                           BB * LL, 3 * HH, WDIM, 0, 0, 0);
    ptr_gemm_f16<<<grid, 128, 0, stream>>>(X16, Wihb16, bih_b, xWb,
                                           BB * LL, 3 * HH, WDIM, 0, 0, 0);
  }

  // --- encoder recurrences ----------------------------------------------
  ptr_zero_f32<<<(BB * HH + 255) / 256, 256, 0, stream>>>(hf0, BB * HH);
  ptr_zero_f32<<<(BB * HH + 255) / 256, 256, 0, stream>>>(hb0, BB * HH);
  ptr_zero_f16<<<(BB * HH + 255) / 256, 256, 0, stream>>>(hf0_16, BB * HH);
  ptr_zero_f16<<<(BB * HH + 255) / 256, 256, 0, stream>>>(hb0_16, BB * HH);

  const size_t sh_enc = (size_t)32 * HH * sizeof(_Float16);   // 32 KB
  for (int t = 0; t < LL; ++t) {                              // forward
    const float*    hin   = (t & 1) ? hf1    : hf0;
    float*          hout  = (t & 1) ? hf0    : hf1;
    const _Float16* hin6  = (t & 1) ? hf1_16 : hf0_16;
    _Float16*       hout6 = (t & 1) ? hf0_16 : hf1_16;
    ptr_gru_step<<<HH / 64, 128, sh_enc, stream>>>(
        hin, hin6, hout, hout6, xWf + (size_t)t * 3 * HH, (long long)LL * 3 * HH,
        Whhf16, bhh_f, enc16 + (size_t)t * DD, (long long)LL * DD, HH);
  }
  for (int s = 0; s < LL; ++s) {                              // backward
    int t = LL - 1 - s;
    const float*    hin   = (s & 1) ? hb1    : hb0;
    float*          hout  = (s & 1) ? hb0    : hb1;
    const _Float16* hin6  = (s & 1) ? hb1_16 : hb0_16;
    _Float16*       hout6 = (s & 1) ? hb0_16 : hb1_16;
    ptr_gru_step<<<HH / 64, 128, sh_enc, stream>>>(
        hin, hin6, hout, hout6, xWb + (size_t)t * 3 * HH, (long long)LL * 3 * HH,
        Whhb16, bhh_b, enc16 + (size_t)t * DD + HH, (long long)LL * DD, HH);
  }
  // L is even -> final states are in hf0 / hb0.

  // --- h0_dec, decoder inputs -------------------------------------------
  ptr_concat_h<<<(BB * DD + 255) / 256, 256, 0, stream>>>(hf0, hb0, hd0, hd0_16);
  ptr_gather_decin<<<BB * KK, 256, 0, stream>>>(enc16, Xidx, decin16);

  // --- decoder input projection -----------------------------------------
  {
    dim3 grid((3 * DD) / 64, (BB * KK) / 64, 1);
    ptr_gemm_f16<<<grid, 128, 0, stream>>>(decin16, Wihd16, bih_d, xWd,
                                           BB * KK, 3 * DD, DD, 0, 0, 0);
  }

  // --- decoder recurrence ------------------------------------------------
  const size_t sh_dec = (size_t)32 * DD * sizeof(_Float16);   // 64 KB
  for (int k = 0; k < KK; ++k) {
    const float*    hin   = (k & 1) ? hd1    : hd0;
    float*          hout  = (k & 1) ? hd0    : hd1;
    const _Float16* hin6  = (k & 1) ? hd1_16 : hd0_16;
    _Float16*       hout6 = (k & 1) ? hd0_16 : hd1_16;
    ptr_gru_step<<<DD / 64, 128, sh_dec, stream>>>(
        hin, hin6, hout, hout6, xWd + (size_t)k * 3 * DD, (long long)KK * 3 * DD,
        Whhd16, bhh_d, dout16 + (size_t)k * DD, (long long)KK * DD, DD);
  }

  // --- attention logits: logits[b,k,l] = dout[b,k,:] . enc[b,l,:] -------
  {
    dim3 grid(LL / 64, KK / 64, BB);
    ptr_gemm_f16<<<grid, 128, 0, stream>>>(
        dout16, enc16, nullptr, logits, KK, LL, DD,
        (long long)KK * DD, (long long)LL * DD, (long long)KK * LL);
  }

  // --- masked log-softmax NLL + mean ------------------------------------
  ptr_nll<<<BB * KK, 256, 0, stream>>>(logits, Xidx, Yidx, nllv);
  ptr_mean<<<1, 256, 0, stream>>>(nllv, (float*)d_out);
}
